// Encoder_67113158967655
// MI455X (gfx1250) — compile-verified, matrix operand order checked by reading, not run on hardware
//
#include <hip/hip_runtime.h>

// ---------------------------------------------------------------------------
// GNN encoder (SAGEConv + ChebConv K=2) for MI455X / gfx1250, wave32.
// Edge scatters are HBM-bound (~1.6 GB traffic); GEMMs use
// v_wmma_f32_16x16x32_bf16 with fp32 accumulation.
// ---------------------------------------------------------------------------

typedef __attribute__((ext_vector_type(16))) __bf16 v16bf;
typedef __attribute__((ext_vector_type(8)))  float  v8f;

#define HID 64

// ------------------------------ zero fill ----------------------------------
__global__ void k_zero(float* __restrict__ p, long long n) {
  long long i = (long long)blockIdx.x * blockDim.x + threadIdx.x;
  if (i < n) p[i] = 0.0f;
}

// ------------------- weight pre-swizzle to WMMA B-fragments ----------------
// Target layout per matrix: frag f = col_tile*2 + kb; per lane a contiguous
// v16bf (32B). Lane holds column n = col_tile*16 + (lane&15); K pattern per
// ISA 16-bit B layout: i<8 -> k = kb*32 + half*8 + i ; i>=8 -> +16.
__global__ void k_swizzle_w(const float* __restrict__ Wl,
                            const float* __restrict__ Wr,
                            const float* __restrict__ W0,
                            const float* __restrict__ W1,
                            __bf16* __restrict__ dst) {
  int idx = blockIdx.x * blockDim.x + threadIdx.x;   // 0..16383
  int mat = idx >> 12;
  int rem = idx & 4095;
  int f    = rem >> 9;          // 0..7
  int lane = (rem >> 4) & 31;
  int i    = rem & 15;
  int col_tile = f >> 1;
  int kb       = f & 1;
  int half     = lane >> 4;
  int kk = (i < 8) ? (half * 8 + i) : (16 + half * 8 + (i - 8));
  int k  = kb * 32 + kk;
  int n  = col_tile * 16 + (lane & 15);
  const float* src = (mat == 0) ? Wl : (mat == 1) ? Wr : (mat == 2) ? W0 : W1;
  dst[idx] = (__bf16)src[k * HID + n];
}

// --------------------------- degree counting -------------------------------
__global__ void k_degrees(const int* __restrict__ ei, int E,
                          float* __restrict__ cnt, float* __restrict__ deg) {
  int e = blockIdx.x * blockDim.x + threadIdx.x;
  if (e >= E) return;
  int s = ei[e];
  int d = ei[E + e];
  atomicAdd(&deg[s], 1.0f);   // degree from src row (Cheb norm)
  atomicAdd(&cnt[d], 1.0f);   // in-count at dst (SAGE mean)
}

// ----------------------- SAGE neighbor scatter-add -------------------------
// one thread per (edge, channel); 64 consecutive threads cover one edge.
__global__ void k_scatter_x(const int* __restrict__ ei, int E,
                            const float* __restrict__ x,
                            float* __restrict__ agg) {
  long long idx = (long long)blockIdx.x * blockDim.x + threadIdx.x;
  int e = (int)(idx >> 6);
  if (e >= E) return;
  int c = (int)(idx & 63);
  int s = ei[e];
  int d = ei[E + e];
  atomicAdd(&agg[(long long)d * HID + c], x[(long long)s * HID + c]);
}

// ------------------------------- mean --------------------------------------
__global__ void k_mean(float* __restrict__ agg, const float* __restrict__ cnt,
                       int Nn) {
  long long idx = (long long)blockIdx.x * blockDim.x + threadIdx.x;
  int node = (int)(idx >> 6);
  if (node >= Nn) return;
  agg[idx] = agg[idx] / fmaxf(cnt[node], 1.0f);
}

// -------------------- A-fragment assembly (fp32 -> bf16) -------------------
// 16-bit A layout 16x32: lane holds row m = lane&15; K values are two
// contiguous 8-runs: {kb*32 + half*8 ..+7} and {+16}.
__device__ __forceinline__ v16bf load_a_frag(const float* __restrict__ A,
                                             int row, int kb, int lane) {
  int half = (lane >> 4) & 1;
  const float* base = A + (long long)row * HID + kb * 32 + half * 8;
  float4 q0 = ((const float4*)base)[0];
  float4 q1 = ((const float4*)base)[1];
  float4 q2 = ((const float4*)(base + 16))[0];
  float4 q3 = ((const float4*)(base + 16))[1];
  v16bf a;
  a[0]  = (__bf16)q0.x; a[1]  = (__bf16)q0.y; a[2]  = (__bf16)q0.z; a[3]  = (__bf16)q0.w;
  a[4]  = (__bf16)q1.x; a[5]  = (__bf16)q1.y; a[6]  = (__bf16)q1.z; a[7]  = (__bf16)q1.w;
  a[8]  = (__bf16)q2.x; a[9]  = (__bf16)q2.y; a[10] = (__bf16)q2.z; a[11] = (__bf16)q2.w;
  a[12] = (__bf16)q3.x; a[13] = (__bf16)q3.y; a[14] = (__bf16)q3.z; a[15] = (__bf16)q3.w;
  return a;
}

// --------------- dual-operand WMMA GEMM: out = A1*B1 + A2*B2 + b -----------
// block = 256 threads = 8 waves; block covers 32 rows x 64 cols,
// one 16x16 output tile per wave; K=64 -> 2 wmma steps per operand.
__global__ void k_gemm_dual(const float* __restrict__ A1,
                            const __bf16* __restrict__ B1s,
                            const float* __restrict__ A2,
                            const __bf16* __restrict__ B2s,
                            const float* __restrict__ bias,
                            const float* __restrict__ pw, int do_prelu,
                            float* __restrict__ out, int Nn) {
  int wave = threadIdx.x >> 5;
  int lane = threadIdx.x & 31;
  int row_tile = wave >> 2;     // 0..1
  int col_tile = wave & 3;      // 0..3
  int row0 = blockIdx.x * 32 + row_tile * 16;
  if (row0 + 16 > Nn) return;
  int row = row0 + (lane & 15);

  const v16bf* B1 = (const v16bf*)B1s;
  const v16bf* B2 = (const v16bf*)B2s;

  v8f acc = {};
#pragma unroll
  for (int kb = 0; kb < 2; ++kb) {
    v16bf a1 = load_a_frag(A1, row, kb, lane);
    v16bf b1 = B1[(col_tile * 2 + kb) * 32 + lane];
    acc = __builtin_amdgcn_wmma_f32_16x16x32_bf16(false, a1, false, b1,
                                                  (short)0, acc, false, false);
    v16bf a2 = load_a_frag(A2, row, kb, lane);
    v16bf b2 = B2[(col_tile * 2 + kb) * 32 + lane];
    acc = __builtin_amdgcn_wmma_f32_16x16x32_bf16(false, a2, false, b2,
                                                  (short)0, acc, false, false);
  }

  // D layout: lane = n within tile; VGPR v -> m = v + 8*(lane>=16)
  int n = col_tile * 16 + (lane & 15);
  float b = bias[n];
  float w = do_prelu ? pw[n] : 0.0f;
  int mhalf = (lane >> 4) * 8;
#pragma unroll
  for (int v = 0; v < 8; ++v) {
    float val = acc[v] + b;
    if (do_prelu) val = (val >= 0.0f) ? val : val * w;
    out[(long long)(row0 + mhalf + v) * HID + n] = val;
  }
}

// ------------- row L2-normalize + PReLU on h, and dis = rsqrt(deg) ---------
__global__ void k_norm_prelu_dis(float* __restrict__ h,
                                 const float* __restrict__ pw,
                                 const float* __restrict__ deg,
                                 float* __restrict__ dis, int Nn) {
  int node = blockIdx.x * blockDim.x + threadIdx.x;
  if (node >= Nn) return;
  float* hp = h + (long long)node * HID;
  float vbuf[HID];
  float ss = 0.0f;
#pragma unroll
  for (int i = 0; i < HID / 4; ++i) {
    float4 q = ((const float4*)hp)[i];
    vbuf[4 * i + 0] = q.x; vbuf[4 * i + 1] = q.y;
    vbuf[4 * i + 2] = q.z; vbuf[4 * i + 3] = q.w;
    ss += q.x * q.x + q.y * q.y + q.z * q.z + q.w * q.w;
  }
  float r = 1.0f / fmaxf(sqrtf(ss), 1e-12f);
#pragma unroll
  for (int c = 0; c < HID; ++c) {
    float v = vbuf[c] * r;
    hp[c] = (v >= 0.0f) ? v : v * pw[c];
  }
  float dg = deg[node];
  dis[node] = (dg > 0.0f) ? rsqrtf(fmaxf(dg, 1.0f)) : 0.0f;
}

// -------------------- Cheb Tx1 weighted scatter-add ------------------------
__global__ void k_scatter_h(const int* __restrict__ ei, int E,
                            const float* __restrict__ h,
                            const float* __restrict__ dis,
                            float* __restrict__ tx1) {
  long long idx = (long long)blockIdx.x * blockDim.x + threadIdx.x;
  int e = (int)(idx >> 6);
  if (e >= E) return;
  int c = (int)(idx & 63);
  int s = ei[e];
  int d = ei[E + e];
  float w = -dis[s] * dis[d];
  atomicAdd(&tx1[(long long)d * HID + c], h[(long long)s * HID + c] * w);
}

// ---------------------------------------------------------------------------
extern "C" void kernel_launch(void* const* d_in, const int* in_sizes, int n_in,
                              void* d_out, int out_size, void* d_ws, size_t ws_size,
                              hipStream_t stream) {
  const float* x   = (const float*)d_in[0];
  const int*   ei  = (const int*)d_in[1];
  const float* W_l = (const float*)d_in[2];
  const float* b_l = (const float*)d_in[3];
  const float* W_r = (const float*)d_in[4];
  const float* W0  = (const float*)d_in[5];
  const float* W1  = (const float*)d_in[6];
  const float* b_c = (const float*)d_in[7];
  const float* pw  = (const float*)d_in[8];
  float* out = (float*)d_out;

  int Nn = in_sizes[0] / HID;   // 100000
  int E  = in_sizes[1] / 2;     // 1600000

  // workspace layout (floats)
  float* ws   = (float*)d_ws;
  float* cnt  = ws;                                   // [N]
  float* deg  = ws + (size_t)Nn;                      // [N]
  float* dis  = ws + 2 * (size_t)Nn;                  // [N]
  float* agg  = ws + 3 * (size_t)Nn;                  // [N,64] -> mean
  float* hbuf = ws + (size_t)Nn * (3 + HID);          // [N,64]
  float* tx1  = ws + (size_t)Nn * (3 + 2 * HID);      // [N,64]
  size_t woff = ((size_t)Nn * (3 + 3 * HID) + 15) & ~(size_t)15; // 64B align
  __bf16* wswz = (__bf16*)(ws + woff);                // 4 x 4096 bf16
  __bf16* Wl_s = wswz;
  __bf16* Wr_s = wswz + 4096;
  __bf16* W0_s = wswz + 8192;
  __bf16* W1_s = wswz + 12288;

  const int T = 256;
  long long nNH = (long long)Nn * HID;
  long long nEH = (long long)E * HID;

  // re-zero atomic accumulators every call (deterministic)
  k_zero<<<(unsigned)((2LL * Nn + T - 1) / T), T, 0, stream>>>(cnt, 2LL * Nn);
  k_zero<<<(unsigned)((nNH + T - 1) / T), T, 0, stream>>>(agg, nNH);
  k_zero<<<(unsigned)((nNH + T - 1) / T), T, 0, stream>>>(tx1, nNH);

  k_swizzle_w<<<64, T, 0, stream>>>(W_l, W_r, W0, W1, wswz);

  k_degrees<<<(unsigned)((E + T - 1) / T), T, 0, stream>>>(ei, E, cnt, deg);
  k_scatter_x<<<(unsigned)((nEH + T - 1) / T), T, 0, stream>>>(ei, E, x, agg);
  k_mean<<<(unsigned)((nNH + T - 1) / T), T, 0, stream>>>(agg, cnt, Nn);

  // h_pre = mean @ W_l + x @ W_r + b_l
  k_gemm_dual<<<(unsigned)((Nn + 31) / 32), T, 0, stream>>>(
      agg, Wl_s, x, Wr_s, b_l, pw, 0, hbuf, Nn);

  k_norm_prelu_dis<<<(unsigned)((Nn + T - 1) / T), T, 0, stream>>>(
      hbuf, pw, deg, dis, Nn);

  k_scatter_h<<<(unsigned)((nEH + T - 1) / T), T, 0, stream>>>(
      ei, E, hbuf, dis, tx1);

  // out = prelu(h @ W0 + Tx1 @ W1 + b_cheb)
  k_gemm_dual<<<(unsigned)((Nn + 31) / 32), T, 0, stream>>>(
      hbuf, W0_s, tx1, W1_s, b_c, pw, 1, out, Nn);
}
